// RNNDecoder_74947179315934
// MI455X (gfx1250) — matrix operand order repaired and checked by reading, not compiled, not run on hardware
//
#include <hip/hip_runtime.h>
#include <hip/hip_bf16.h>
#include <cstdint>
#include <cstddef>

// ---------------------------------------------------------------------------
// Graph-LSTM decoder for MI455X (gfx1250), wave32 + WMMA bf16 path.
// B=8, N=512, T=64, NIN=64, NEMB=NH=128.
// All GEMMs use v_wmma_f32_16x16x32_bf16 (bf16 in, fp32 accum).
// LSTM state c, gate preacts, and residual Xi stay fp32.
// z-staging (es|h -> LDS) uses GLOBAL_LOAD_ASYNC_TO_LDS_B128, hidden behind
// the K=512 An@h WMMA phase; completion gated by s_wait_asynccnt.
// ---------------------------------------------------------------------------

typedef __attribute__((ext_vector_type(16))) __bf16        v16bf;
typedef __attribute__((ext_vector_type(8)))  float         v8f;
typedef __attribute__((ext_vector_type(4)))  unsigned int  v4u;

#define BB   8
#define NN   512
#define TT   64
#define NINF 64
#define NH   128
#define ZD   384   // [es(128) | eh(128) | h(128)]

__device__ inline v8f vzero8() {
  v8f z;
#pragma unroll
  for (int i = 0; i < 8; ++i) z[i] = 0.0f;
  return z;
}

__device__ inline v8f wmma_bf16(v16bf a, v16bf b, v8f c) {
  // v_wmma_f32_16x16x32_bf16  D = A(16x32) * B(32x16) + C(16x16 f32)
  return __builtin_amdgcn_wmma_f32_16x16x32_bf16(
      false, a, false, b, (short)0, c, false, false);
}

// A-fragment (16x32 bf16) from a ROW-MAJOR tile, leading dim `ld` elements.
// ISA layout (7.12.2): lane r=lane&15 holds row M=r; half=lane>>4:
//   elems 0..7  = K  8*half + [0..7]
//   elems 8..15 = K 16 + 8*half + [0..7]
__device__ inline v16bf a_frag(const __bf16* base, int ld, int lane) {
  const int r = lane & 15, hf = lane >> 4;
  const __bf16* p = base + (size_t)r * ld + 8 * hf;
  union { v16bf v; v4u u[2]; } x;
  x.u[0] = *(const v4u*)(p);
  x.u[1] = *(const v4u*)(p + 16);
  return x.v;
}

// B-fragment (32x16 bf16) from a COLUMN-MAJOR store Bt[col][k], ld = K stride.
// ISA layout: lane c=lane&15 is column N=c; half=lane>>4:
//   elems 0..7 = K 8*half + [0..7], elems 8..15 = K 16 + 8*half + [0..7] (matches a_frag).
__device__ inline v16bf b_frag(const __bf16* bt, int ld, int lane) {
  const int c = lane & 15, hf = lane >> 4;
  const __bf16* p = bt + (size_t)c * ld + 8 * hf;
  union { v16bf v; v4u u[2]; } x;
  x.u[0] = *(const v4u*)(p);
  x.u[1] = *(const v4u*)(p + 16);
  return x.v;
}

// Async DMA one 16-byte chunk global -> LDS (tracked by ASYNCcnt).
__device__ inline void async_g2l_b128(void* lds_ptr, const void* gptr) {
#if defined(__gfx1250__)
  const unsigned lds_addr = (unsigned)(size_t)lds_ptr;           // addr[31:0] = LDS byte addr
  const unsigned long long ga = (unsigned long long)(size_t)gptr;
  asm volatile("global_load_async_to_lds_b128 %0, %1, off"
               :: "v"(lds_addr), "v"(ga)
               : "memory");
#else
  // Fallback for non-gfx1250 device passes: synchronous copy.
  *(v4u*)lds_ptr = *(const v4u*)gptr;
#endif
}

__device__ inline void wait_async0() {
#if defined(__gfx1250__)
  asm volatile("s_wait_asynccnt 0x0" ::: "memory");
#endif
}

__device__ inline float sigf(float x) { return 1.0f / (1.0f + __expf(-x)); }

// ---------------------------------------------------------------------------
// Setup kernels
// ---------------------------------------------------------------------------

struct PrepArgs {
  const float *Wse, *Wpe, *Wout;
  const float *wx0, *wx1, *wx2, *wx3;  // Wii, Wif, Wig, Wio   (256x128)
  const float *wh0, *wh1, *wh2, *wh3;  // Whi, Whf, Whg, Who   (128x128)
  const float *bx0, *bx1, *bx2, *bx3;
  const float *bh0, *bh1, *bh2, *bh3;
  __bf16 *WseT, *WpeT, *WoutT, *WgT;
  float* bg;
};

__global__ void k_prep(PrepArgs p) {
  const int i0 = blockIdx.x * blockDim.x + threadIdx.x;
  const int stride = gridDim.x * blockDim.x;
  // Folded gate weights, transposed: WgT[o=gate*128+oo][k=0..383]
  for (int idx = i0; idx < 512 * ZD; idx += stride) {
    const int o = idx / ZD, k = idx % ZD;
    const int gate = o >> 7, oo = o & 127;
    const float* wx = (gate == 0) ? p.wx0 : (gate == 1) ? p.wx1 : (gate == 2) ? p.wx2 : p.wx3;
    const float* wh = (gate == 0) ? p.wh0 : (gate == 1) ? p.wh1 : (gate == 2) ? p.wh2 : p.wh3;
    const float v = (k < 256) ? wx[(size_t)k * 128 + oo] : wh[(size_t)(k - 256) * 128 + oo];
    p.WgT[idx] = (__bf16)v;
  }
  for (int idx = i0; idx < 128 * 128; idx += stride) {   // WpeT[o][k], Wpe is 128x128
    const int o = idx >> 7, k = idx & 127;
    p.WpeT[idx] = (__bf16)p.Wpe[(size_t)k * 128 + o];
  }
  for (int idx = i0; idx < 64 * 128; idx += stride) {    // WoutT[o(64)][k(128)], Wout 128x64
    const int o = idx >> 7, k = idx & 127;
    p.WoutT[idx] = (__bf16)p.Wout[(size_t)k * 64 + o];
  }
  for (int idx = i0; idx < 128 * 64; idx += stride) {    // WseT[o(128)][k(64)], Wse 64x128
    const int o = idx >> 6, k = idx & 63;
    p.WseT[idx] = (__bf16)p.Wse[(size_t)k * 128 + o];
  }
  for (int idx = i0; idx < 512; idx += stride) {         // folded gate bias
    const int gate = idx >> 7, oo = idx & 127;
    const float* bx = (gate == 0) ? p.bx0 : (gate == 1) ? p.bx1 : (gate == 2) ? p.bx2 : p.bx3;
    const float* bh = (gate == 0) ? p.bh0 : (gate == 1) ? p.bh1 : (gate == 2) ? p.bh2 : p.bh3;
    p.bg[idx] = bx[oo] + bh[oo];
  }
}

// dinv[row] = d>0 ? 1/sqrt(d) : 0 ; one wave per row of A
__global__ void k_dinv(const float* __restrict__ A, float* __restrict__ dinv) {
  const int row  = blockIdx.x * 8 + (threadIdx.x >> 5);  // 4096 rows
  const int lane = threadIdx.x & 31;
  const float* ar = A + (size_t)row * NN;
  float s = 0.0f;
  for (int j = lane; j < NN; j += 32) s += ar[j];
#pragma unroll
  for (int off = 16; off; off >>= 1) s += __shfl_down(s, off, 32);
  if (lane == 0) dinv[row] = (s > 0.0f) ? rsqrtf(s) : 0.0f;
}

__global__ void k_An(const float* __restrict__ A, const float* __restrict__ dinv,
                     __bf16* __restrict__ An) {
  const size_t total = (size_t)BB * NN * NN;
  const size_t stride = (size_t)gridDim.x * blockDim.x;
  for (size_t idx = (size_t)blockIdx.x * blockDim.x + threadIdx.x; idx < total; idx += stride) {
    const int m = (int)(idx & 511);
    const int n = (int)((idx >> 9) & 511);
    const int b = (int)(idx >> 18);
    An[idx] = (__bf16)(A[idx] * dinv[b * NN + n] * dinv[b * NN + m]);
  }
}

// ---------------------------------------------------------------------------
// Init: es = X0 @ Wse + bse (WMMA), zero h/c, Xi = X0, out[:, :, 0, :] = X0
// One block = 16 nodes, 8 waves (wave w owns output features [16w,16w+16)).
// ---------------------------------------------------------------------------

struct InitArgs {
  const float* X;        // [8][512][64][64]
  const float* bse;      // [128]
  const __bf16* WseT;    // [128][64]
  __bf16* es;            // [8][512][128]
  __bf16* hrm;           // [8][512][128]
  __bf16* ht;            // [8][128][512]
  float* c;              // [8][512][128]
  float* Xi;             // [8][512][64]
  float* out;            // [8][512][64][64]
};

__global__ __launch_bounds__(256) void k_init(InitArgs s) {
  __shared__ __align__(16) __bf16 x0s[16 * NINF];
  const int tid  = threadIdx.x;
  const int lane = tid & 31;
  const int wave = tid >> 5;
  const int b    = blockIdx.x >> 5;
  const int m0   = (blockIdx.x & 31) << 4;
  const int node0 = b * NN + m0;

  for (int idx = tid; idx < 16 * NINF; idx += 256) {
    const int rr = idx >> 6, j = idx & 63;
    const float v = s.X[(((size_t)(node0 + rr)) * TT + 0) * NINF + j];
    x0s[rr * NINF + j] = (__bf16)v;
    s.Xi[(size_t)(node0 + rr) * NINF + j] = v;
    s.out[(((size_t)(node0 + rr)) * TT + 0) * NINF + j] = v;
  }
  for (int idx = tid; idx < 16 * NH; idx += 256) {
    const int rr = idx >> 7, f = idx & 127;
    s.c[(size_t)(node0 + rr) * NH + f]            = 0.0f;
    s.hrm[(size_t)(node0 + rr) * NH + f]          = (__bf16)0.0f;
    s.ht[((size_t)b * NH + f) * NN + (m0 + rr)]   = (__bf16)0.0f;
  }
  __syncthreads();

  const int f0 = wave << 4;
  v8f acc = vzero8();
#pragma unroll
  for (int k0 = 0; k0 < NINF; k0 += 32) {
    v16bf a = a_frag(x0s + k0, NINF, lane);
    v16bf bm = b_frag(s.WseT + (size_t)f0 * NINF + k0, NINF, lane);
    acc = wmma_bf16(a, bm, acc);
  }
  const int r = lane & 15, hf = lane >> 4;
  const float bv = s.bse[f0 + r];
#pragma unroll
  for (int i = 0; i < 8; ++i)
    s.es[(size_t)(node0 + i + 8 * hf) * NH + f0 + r] = (__bf16)(acc[i] + bv);
}

// ---------------------------------------------------------------------------
// One fused recurrence step: H=An@h -> eh=H@Wpe -> 4 gates -> LSTM cell ->
// Xi += h@Wout, write out[:, :, t, :].  Block = 16 nodes, 8 waves.
// ---------------------------------------------------------------------------

struct StepArgs {
  const __bf16* An;     // [8][512][512]
  const __bf16* es;     // [8][512][128]
  __bf16* hrm;          // [8][512][128]  row-major h (A-side)
  __bf16* ht;           // [8][128][512]  transposed h (B-side of An@h)
  float* c;             // [8][512][128]
  float* Xi;            // [8][512][64]
  const __bf16* WpeT;   // [128][128]
  const __bf16* WgT;    // [512][384]
  const __bf16* WoutT;  // [64][128]
  const float* bg;      // [512]
  const float* bpe;     // [128]
  const float* bout;    // [64]
  float* out;           // [8][512][64][64]
  int t;
};

__global__ __launch_bounds__(256) void k_step(StepArgs s) {
  __shared__ __align__(16) __bf16 zs[16 * ZD];   // z = [es | eh | h] per node row
  __shared__ __align__(16) __bf16 Hs[16 * NH];   // H = An@h tile
  __shared__ __align__(16) __bf16 hn[16 * NH];   // h_new tile
  const int tid  = threadIdx.x;
  const int lane = tid & 31;
  const int wave = tid >> 5;
  const int b    = blockIdx.x >> 5;
  const int m0   = (blockIdx.x & 31) << 4;
  const int node0 = b * NN + m0;
  const int f0   = wave << 4;
  const int rcol = lane & 15, hf = lane >> 4;

  // Async-stage the es / h parts of z into LDS; the DMA is hidden behind the
  // Phase-1 WMMA chain and only waited on before the Phase-3 barrier.
  // 256 threads x one 16B chunk each exactly covers 16 rows x 128 bf16 cols.
  {
    const int rr = tid >> 4;          // row 0..15
    const int cc = (tid & 15) << 3;   // col chunk base (8 bf16 = 16 B)
    async_g2l_b128(&zs[rr * ZD + cc],
                   s.es + (size_t)(node0 + rr) * NH + cc);
    async_g2l_b128(&zs[rr * ZD + 256 + cc],
                   s.hrm + (size_t)(node0 + rr) * NH + cc);
  }

  // ---- Phase 1: H = An @ h   (M=16 nodes, K=512 neighbors, wave-local N=16)
  {
    v8f acc = vzero8();
    const __bf16* Ab = s.An + ((size_t)b * NN + m0) * NN;
    const __bf16* Bt = s.ht + ((size_t)b * NH + f0) * NN;
#pragma unroll
    for (int k0 = 0; k0 < NN; k0 += 32) {
      v16bf a  = a_frag(Ab + k0, NN, lane);
      v16bf bm = b_frag(Bt + k0, NN, lane);
      acc = wmma_bf16(a, bm, acc);
    }
#pragma unroll
    for (int i = 0; i < 8; ++i)
      Hs[(i + 8 * hf) * NH + f0 + rcol] = (__bf16)acc[i];
  }
  __syncthreads();

  // ---- Phase 2: eh = H @ Wpe + bpe   (K=128)
  {
    v8f acc = vzero8();
#pragma unroll
    for (int k0 = 0; k0 < NH; k0 += 32) {
      v16bf a  = a_frag(Hs + k0, NH, lane);
      v16bf bm = b_frag(s.WpeT + (size_t)f0 * NH + k0, NH, lane);
      acc = wmma_bf16(a, bm, acc);
    }
    const float bv = s.bpe[f0 + rcol];
#pragma unroll
    for (int i = 0; i < 8; ++i)
      zs[(i + 8 * hf) * ZD + 128 + f0 + rcol] = (__bf16)(acc[i] + bv);
  }
  wait_async0();        // es/h DMA into zs must land before phase 3 reads
  __syncthreads();

  // ---- Phase 3: gate preacts = z @ WgT^T, then LSTM cell update.
  // Wave w computes feature tile f0 for all 4 gates (gate gi -> cols gi*128+f0).
  {
    v8f ag[4];
#pragma unroll
    for (int gi = 0; gi < 4; ++gi) {
      v8f acc = vzero8();
      const __bf16* wb = s.WgT + (size_t)(gi * NH + f0) * ZD;
#pragma unroll
      for (int k0 = 0; k0 < ZD; k0 += 32) {
        v16bf a  = a_frag(zs + k0, ZD, lane);
        v16bf bm = b_frag(wb + k0, ZD, lane);
        acc = wmma_bf16(a, bm, acc);
      }
      ag[gi] = acc;
    }
    const int feat = f0 + rcol;
    const float b_i = s.bg[0 * NH + feat];
    const float b_f = s.bg[1 * NH + feat];
    const float b_g = s.bg[2 * NH + feat];
    const float b_o = s.bg[3 * NH + feat];
#pragma unroll
    for (int i = 0; i < 8; ++i) {
      const int m = i + 8 * hf;
      const size_t ci = (size_t)(node0 + m) * NH + feat;
      const float co = s.c[ci];
      const float iv = sigf(ag[0][i] + b_i);
      const float fv = sigf(ag[1][i] + b_f);
      const float gv = tanhf(ag[2][i] + b_g);
      const float ov = sigf(ag[3][i] + b_o);
      const float cn = fv * co + iv * gv;
      const float hv = ov * tanhf(cn);
      s.c[ci] = cn;
      const __bf16 hb = (__bf16)hv;
      s.hrm[ci] = hb;
      s.ht[((size_t)b * NH + feat) * NN + (m0 + m)] = hb;
      hn[m * NH + feat] = hb;
    }
  }
  __syncthreads();

  // ---- Phase 4: Xi += h_new @ Wout + bout ; emit out[:, :, t, :]. 4 waves.
  if (wave < 4) {
    const int o0 = wave << 4;
    v8f acc = vzero8();
#pragma unroll
    for (int k0 = 0; k0 < NH; k0 += 32) {
      v16bf a  = a_frag(hn + k0, NH, lane);
      v16bf bm = b_frag(s.WoutT + (size_t)o0 * NH + k0, NH, lane);
      acc = wmma_bf16(a, bm, acc);
    }
    const int col = o0 + rcol;
    const float bv = s.bout[col];
#pragma unroll
    for (int i = 0; i < 8; ++i) {
      const int m = i + 8 * hf;
      const size_t xi = (size_t)(node0 + m) * NINF + col;
      const float xv = s.Xi[xi] + acc[i] + bv;
      s.Xi[xi] = xv;
      s.out[(((size_t)(node0 + m)) * TT + s.t) * NINF + col] = xv;
    }
  }
}

// ---------------------------------------------------------------------------
// Host launcher
// ---------------------------------------------------------------------------

extern "C" void kernel_launch(void* const* d_in, const int* in_sizes, int n_in,
                              void* d_out, int out_size, void* d_ws, size_t ws_size,
                              hipStream_t stream) {
  (void)in_sizes; (void)n_in; (void)out_size; (void)ws_size;
  const float* X    = (const float*)d_in[0];
  const float* A    = (const float*)d_in[1];
  const float* Wse  = (const float*)d_in[2];
  const float* bse  = (const float*)d_in[3];
  const float* Wpe  = (const float*)d_in[4];
  const float* bpe  = (const float*)d_in[5];
  const float* Wii  = (const float*)d_in[6];
  const float* bii  = (const float*)d_in[7];
  const float* Whi  = (const float*)d_in[8];
  const float* bhi  = (const float*)d_in[9];
  const float* Wif  = (const float*)d_in[10];
  const float* bif  = (const float*)d_in[11];
  const float* Whf  = (const float*)d_in[12];
  const float* bhf  = (const float*)d_in[13];
  const float* Wig  = (const float*)d_in[14];
  const float* big  = (const float*)d_in[15];
  const float* Whg  = (const float*)d_in[16];
  const float* bhg  = (const float*)d_in[17];
  const float* Wio  = (const float*)d_in[18];
  const float* bio  = (const float*)d_in[19];
  const float* Who  = (const float*)d_in[20];
  const float* bho  = (const float*)d_in[21];
  const float* Wout = (const float*)d_in[22];
  const float* bout = (const float*)d_in[23];
  float* out = (float*)d_out;

  // Workspace carve-out (256 B aligned); total ~11 MB.
  char* w = (char*)d_ws;
  auto alloc = [&](size_t bytes) -> char* {
    char* p = w;
    w += (bytes + 255) & ~(size_t)255;
    return p;
  };
  __bf16* An    = (__bf16*)alloc((size_t)BB * NN * NN * 2);
  float*  dinv  = (float*) alloc((size_t)BB * NN * 4);
  __bf16* ht    = (__bf16*)alloc((size_t)BB * NH * NN * 2);
  __bf16* hrm   = (__bf16*)alloc((size_t)BB * NN * NH * 2);
  float*  cst   = (float*) alloc((size_t)BB * NN * NH * 4);
  __bf16* es    = (__bf16*)alloc((size_t)BB * NN * NH * 2);
  float*  Xi    = (float*) alloc((size_t)BB * NN * NINF * 4);
  __bf16* WgT   = (__bf16*)alloc((size_t)512 * ZD * 2);
  __bf16* WpeT  = (__bf16*)alloc((size_t)NH * NH * 2);
  __bf16* WoutT = (__bf16*)alloc((size_t)NINF * NH * 2);
  __bf16* WseT  = (__bf16*)alloc((size_t)NH * NINF * 2);
  float*  bg    = (float*) alloc((size_t)512 * 4);

  PrepArgs pa{Wse, Wpe, Wout,
              Wii, Wif, Wig, Wio,
              Whi, Whf, Whg, Who,
              bii, bif, big, bio,
              bhi, bhf, bhg, bho,
              WseT, WpeT, WoutT, WgT, bg};
  k_prep<<<256, 256, 0, stream>>>(pa);
  k_dinv<<<(BB * NN) / 8, 256, 0, stream>>>(A, dinv);
  k_An<<<1024, 256, 0, stream>>>(A, dinv, An);

  InitArgs ia{X, bse, WseT, es, hrm, ht, cst, Xi, out};
  k_init<<<(BB * NN) / 16, 256, 0, stream>>>(ia);

  StepArgs sa{An, es, hrm, ht, cst, Xi, WpeT, WgT, WoutT, bg, bpe, bout, out, 0};
  for (int t = 1; t < TT; ++t) {
    sa.t = t;
    k_step<<<(BB * NN) / 16, 256, 0, stream>>>(sa);
  }
}
